// Attention_41850161332702
// MI455X (gfx1250) — matrix operand order repaired and checked by reading, not compiled, not run on hardware
//
#include <hip/hip_runtime.h>
#include <math.h>

// Problem constants (match reference: B=64, T=4096, D=512)
#define BB 64
#define TT 4096
#define DD 512
#define TCHUNKS 8
#define ROWS_PER_CHUNK (TT / TCHUNKS)   // 512 rows of attend_to per block
#define WAVES_PER_BLOCK 8               // 256 threads / wave32

typedef __attribute__((ext_vector_type(16))) _Float16 v16h;
typedef __attribute__((ext_vector_type(8)))  float    v8f;

// ---------------------------------------------------------------------------
// Kernel 1: k = query @ W^T + bias   (64x512 = 4x32 tiles of 16x16)
// One wave per 16x16 output tile; K-loop of 16 x v_wmma_f32_16x16x32_f16.
// A fragment (16x32 f16): lane l holds row M=(l&15); elem e = K = h*16+e,
// where h = l>>4 (same k order as B).
// B fragment (32x16 f16): lane l holds col N=(l&15); elem e = B[K=h*16+e][N]
// with B[k][n] = W[n][k] (row-major W -> contiguous f32 loads, cvt to f16).
// C/D (16x16 f32, v8f): row r maps to M = r + 8*h, N = l&15.
// ---------------------------------------------------------------------------
__global__ void linear_wmma(const float* __restrict__ query,
                            const float* __restrict__ W,
                            const float* __restrict__ bias,
                            float* __restrict__ kvec)
{
    const int tileM = blockIdx.x >> 5;   // 0..3
    const int tileN = blockIdx.x & 31;   // 0..31
    const int m0 = tileM * 16;
    const int n0 = tileN * 16;
    const int lane = threadIdx.x & 31;
    const int h  = lane >> 4;
    const int mr = lane & 15;

    v8f c = {};
    for (int k0 = 0; k0 < DD; k0 += 32) {
        v16h a, bm;
        const float* qrow = query + (m0 + mr) * DD + k0;
#pragma unroll
        for (int e = 0; e < 8; ++e) {
            a[e]     = (_Float16)qrow[h * 16 + e];
            a[8 + e] = (_Float16)qrow[h * 16 + 8 + e];
        }
        const float* wrow = W + (n0 + mr) * DD + k0 + h * 16;
#pragma unroll
        for (int e = 0; e < 16; ++e)
            bm[e] = (_Float16)wrow[e];
        // (neg_a, A, neg_b, B, c_mod, C, reuse_a, reuse_b)
        c = __builtin_amdgcn_wmma_f32_16x16x32_f16(false, a, false, bm,
                                                   (short)0, c, false, false);
    }
    const float bn = bias[n0 + mr];
#pragma unroll
    for (int r = 0; r < 8; ++r)
        kvec[(m0 + r + 8 * h) * DD + n0 + mr] = c[r] + bn;
}

// ---------------------------------------------------------------------------
// Kernel 2: fused streaming scores + online softmax + weighted accumulation.
// Grid (TCHUNKS, B); 256 threads = 8 waves. Each wave owns 64 rows (stride 8),
// each lane owns d-slots {j*128 + lane*4 .. +3 | j=0..3} (16 floats).
// Masked rows (mask[t*B+b] != 0) are skipped BEFORE loading -> ~half traffic.
// Per-wave running (m, s, ctx) merged across waves through LDS; partials
// (m, s, ctx[512]) per (b, chunk) written to workspace.
// ---------------------------------------------------------------------------
__global__ void attn_stream(const float* __restrict__ attend_to,
                            const unsigned char* __restrict__ mask,
                            const float* __restrict__ kvec,
                            float* __restrict__ part_ctx,
                            float* __restrict__ part_m,
                            float* __restrict__ part_s)
{
    const int b     = blockIdx.y;
    const int chunk = blockIdx.x;
    const int wave  = threadIdx.x >> 5;
    const int lane  = threadIdx.x & 31;
    const int t0    = chunk * ROWS_PER_CHUNK;
    const float* Ab = attend_to + (size_t)b * TT * DD;

    float4 kv[4];
#pragma unroll
    for (int j = 0; j < 4; ++j)
        kv[j] = *(const float4*)(kvec + b * DD + j * 128 + lane * 4);

    float m = -INFINITY, ssum = 0.f;
    float4 ctx[4];
#pragma unroll
    for (int j = 0; j < 4; ++j) ctx[j] = make_float4(0.f, 0.f, 0.f, 0.f);

    const int iters = ROWS_PER_CHUNK / WAVES_PER_BLOCK;  // 64
    for (int i = 0; i < iters; ++i) {
        const int t = t0 + i * WAVES_PER_BLOCK + wave;
        // Prefetch next row this wave will touch (32 lanes * 64B = full 2KB row),
        // but only if it is unmasked (masked rows are never loaded).
        if (i + 1 < iters) {
            const int tn = t + WAVES_PER_BLOCK;
            if (!mask[(size_t)tn * BB + b])
                __builtin_prefetch(Ab + (size_t)tn * DD + lane * 16, 0, 0);
        }
        if (mask[(size_t)t * BB + b]) continue;  // True = masked out (weight 0)

        const float* row = Ab + (size_t)t * DD;
        float4 r[4];
#pragma unroll
        for (int j = 0; j < 4; ++j)
            r[j] = *(const float4*)(row + j * 128 + lane * 4);

        float dot = 0.f;
#pragma unroll
        for (int j = 0; j < 4; ++j)
            dot += r[j].x * kv[j].x + r[j].y * kv[j].y +
                   r[j].z * kv[j].z + r[j].w * kv[j].w;
#pragma unroll
        for (int off = 16; off > 0; off >>= 1)
            dot += __shfl_xor(dot, off, 32);

        // online softmax update
        const float mn   = fmaxf(m, dot);
        const float corr = __expf(m - mn);   // exp(-inf)=0 handles first row
        const float w    = __expf(dot - mn);
        m = mn;
        ssum = ssum * corr + w;
#pragma unroll
        for (int j = 0; j < 4; ++j) {
            ctx[j].x = ctx[j].x * corr + w * r[j].x;
            ctx[j].y = ctx[j].y * corr + w * r[j].y;
            ctx[j].z = ctx[j].z * corr + w * r[j].z;
            ctx[j].w = ctx[j].w * corr + w * r[j].w;
        }
    }

    // merge the 8 waves of this block through LDS
    __shared__ float lm[WAVES_PER_BLOCK];
    __shared__ float ls[WAVES_PER_BLOCK];
    __shared__ __align__(16) float lctx[WAVES_PER_BLOCK][DD];
    if (lane == 0) { lm[wave] = m; ls[wave] = ssum; }
#pragma unroll
    for (int j = 0; j < 4; ++j)
        *(float4*)&lctx[wave][j * 128 + lane * 4] = ctx[j];
    __syncthreads();

    float M = -INFINITY;
#pragma unroll
    for (int w2 = 0; w2 < WAVES_PER_BLOCK; ++w2) M = fmaxf(M, lm[w2]);
    float ew[WAVES_PER_BLOCK];
    float S = 0.f;
#pragma unroll
    for (int w2 = 0; w2 < WAVES_PER_BLOCK; ++w2) {
        ew[w2] = (lm[w2] == -INFINITY) ? 0.f : __expf(lm[w2] - M);
        S += ew[w2] * ls[w2];
    }
    const int pbase = (b * TCHUNKS + chunk) * DD;
    for (int d = threadIdx.x; d < DD; d += blockDim.x) {
        float acc = 0.f;
#pragma unroll
        for (int w2 = 0; w2 < WAVES_PER_BLOCK; ++w2)
            acc += ew[w2] * lctx[w2][d];
        part_ctx[pbase + d] = acc;
    }
    if (threadIdx.x == 0) {
        part_m[b * TCHUNKS + chunk] = M;
        part_s[b * TCHUNKS + chunk] = S;
    }
}

// ---------------------------------------------------------------------------
// Kernel 3: combine the TCHUNKS partials per batch row and normalize.
// ---------------------------------------------------------------------------
__global__ void attn_combine(const float* __restrict__ part_ctx,
                             const float* __restrict__ part_m,
                             const float* __restrict__ part_s,
                             float* __restrict__ out)
{
    const int b = blockIdx.x;
    float M = -INFINITY;
#pragma unroll
    for (int w = 0; w < TCHUNKS; ++w) M = fmaxf(M, part_m[b * TCHUNKS + w]);
    float ew[TCHUNKS];
    float S = 0.f;
#pragma unroll
    for (int w = 0; w < TCHUNKS; ++w) {
        const float mw = part_m[b * TCHUNKS + w];
        ew[w] = (mw == -INFINITY) ? 0.f : __expf(mw - M);
        S += ew[w] * part_s[b * TCHUNKS + w];
    }
    const float inv = 1.f / S;
    for (int d = threadIdx.x; d < DD; d += blockDim.x) {
        float acc = 0.f;
#pragma unroll
        for (int w = 0; w < TCHUNKS; ++w)
            acc += ew[w] * part_ctx[(b * TCHUNKS + w) * DD + d];
        out[b * DD + d] = acc * inv;
    }
}

extern "C" void kernel_launch(void* const* d_in, const int* in_sizes, int n_in,
                              void* d_out, int out_size, void* d_ws, size_t ws_size,
                              hipStream_t stream) {
    (void)in_sizes; (void)n_in; (void)out_size; (void)ws_size;
    const float*         query     = (const float*)d_in[0];          // [B, D]
    const float*         attend_to = (const float*)d_in[1];          // [B, T, D]
    const unsigned char* mask      = (const unsigned char*)d_in[2];  // [T, B] bool
    const float*         W         = (const float*)d_in[3];          // [D, D]
    const float*         bias      = (const float*)d_in[4];          // [D]
    float*               out       = (float*)d_out;                  // [B, 1, D]

    // workspace layout (floats): kvec[B*D] | part_ctx[B*TCHUNKS*D] | part_m | part_s
    float* kvec     = (float*)d_ws;
    float* part_ctx = kvec + BB * DD;
    float* part_m   = part_ctx + BB * TCHUNKS * DD;
    float* part_s   = part_m + BB * TCHUNKS;

    linear_wmma<<<dim3((BB / 16) * (DD / 16)), dim3(32), 0, stream>>>(query, W, bias, kvec);
    attn_stream<<<dim3(TCHUNKS, BB), dim3(32 * WAVES_PER_BLOCK), 0, stream>>>(
        attend_to, mask, kvec, part_ctx, part_m, part_s);
    attn_combine<<<dim3(BB), dim3(128), 0, stream>>>(part_ctx, part_m, part_s, out);
}
